// Rvt_7344394076609
// MI455X (gfx1250) — matrix operand order repaired and checked by reading, not compiled, not run on hardware
//
#include <hip/hip_runtime.h>
#include <stdint.h>

// ---------------- problem constants ----------------
static constexpr int N_    = 16;
static constexpr int C_    = 256;   // channels everywhere (R=A=P=O=256)
static constexpr int H_    = 32;
static constexpr int W_    = 32;
static constexpr int HW_   = 1024;
static constexpr int DV_   = 900;   // 30*30 valid-conv tokens
static constexpr int DPAD_ = 928;   // padded to multiple of 32
static constexpr int HEADS_= 8;
static constexpr int HC_   = 32;    // channels per head
static constexpr int K3_   = 2304;  // 9*256 implicit-GEMM K for 3x3 convs

typedef __attribute__((ext_vector_type(16))) __bf16 v16bf;
typedef __attribute__((ext_vector_type(8)))  float  v8f;

// TDM availability (arity differs: ROCm7.2/clang-22 = 5 args, clang-23 = 6 args)
#if defined(__HIP_DEVICE_COMPILE__) && \
    __has_builtin(__builtin_amdgcn_tensor_load_to_lds) && \
    __has_builtin(__builtin_amdgcn_s_wait_tensorcnt)
#define HAVE_TDM 1
#else
#define HAVE_TDM 0
#endif

// ---------------- helpers ----------------
__device__ __forceinline__ unsigned f2bf1(float x) {
  union { float f; unsigned u; } cv; cv.f = x;
  unsigned u = cv.u;
  return (u + 0x7fffu + ((u >> 16) & 1u)) >> 16;   // round-to-nearest-even
}
__device__ __forceinline__ unsigned short f2bf(float x) { return (unsigned short)f2bf1(x); }
__device__ __forceinline__ unsigned pk2(float lo, float hi) {
  return f2bf1(lo) | (f2bf1(hi) << 16);
}
// pack 8 f32 (scaled) -> 8 bf16 in a uint4
__device__ __forceinline__ uint4 pk8(const v8f a, float s) {
  uint4 r;
  r.x = pk2(a[0] * s, a[1] * s);
  r.y = pk2(a[2] * s, a[3] * s);
  r.z = pk2(a[4] * s, a[5] * s);
  r.w = pk2(a[6] * s, a[7] * s);
  return r;
}
__device__ __forceinline__ float tanh_fast(float x) {       // branch-free
  return 1.0f - 2.0f / (1.0f + __expf(2.0f * x));
}
__device__ __forceinline__ float sigm_fast(float x) {
  return 1.0f / (1.0f + __expf(-x));
}

// A fragment (16x32 MxK bf16) from LDS tile [16 rows][rowStride], K contiguous.
// lane L: row = L%16, half h = L/16; VGPR v<4: K=8h+2v{,+1}; v>=4: K=16+8h+2(v-4){,+1}.
__device__ __forceinline__ v16bf frag_a_lds_s(const unsigned short* s, int rowStride) {
  int l = threadIdx.x & 31;
  int r = l & 15, h = l >> 4;
  union { v16bf v; unsigned u[8]; } f;
#pragma unroll
  for (int vv = 0; vv < 8; ++vv) {
    int K = ((vv < 4) ? 0 : 16) + 8 * h + 2 * (vv & 3);
    f.u[vv] = *(const unsigned*)(s + r * rowStride + K);
  }
  return f.v;
}
__device__ __forceinline__ v16bf frag_a_lds(const unsigned short* s) {
  return frag_a_lds_s(s, 32);
}

// B fragment (32x16 KxN bf16) from LDS tile [16 cols][32 K], K contiguous per col.
// lane L: col = L%16, half h = L/16; VGPR v<4: K=8h+2v{,+1}; v>=4: K=16+8h+2(v-4){,+1}.
__device__ __forceinline__ v16bf frag_b_lds(const unsigned short* s) {
  int l = threadIdx.x & 31;
  int c = l & 15, h = l >> 4;
  union { v16bf v; unsigned u[8]; } f;
#pragma unroll
  for (int vv = 0; vv < 8; ++vv) {
    int K = ((vv < 4) ? 0 : 16) + 8 * h + 2 * (vv & 3);
    f.u[vv] = *(const unsigned*)(s + c * 32 + K);
  }
  return f.v;
}

__device__ __forceinline__ v8f wmma_bf16(v16bf a, v16bf b, v8f c) {
  return __builtin_amdgcn_wmma_f32_16x16x32_bf16(false, a, false, b, (short)0, c,
                                                 false, false);
}

#if HAVE_TDM
typedef __attribute__((ext_vector_type(4))) unsigned tdm_u32x4;
typedef __attribute__((ext_vector_type(8))) int      tdm_i32x8;
typedef __attribute__((ext_vector_type(4))) int      tdm_i32x4;

// 1-D contiguous copy of nelem 2-byte elements, global -> LDS, via the TDM.
// D# per CDNA5 ISA (group0: count/lds/global/type,
// group1: data_size=2B, tensor_dim0=tile_dim0=nelem, stride=nelem).
__device__ __forceinline__ void tdm_load_1d(unsigned lds_off, const void* gptr,
                                            unsigned nelem) {
  unsigned long long ga = (unsigned long long)(uintptr_t)gptr;
  tdm_u32x4 g0;
  g0[0] = 1u;                                           // count=1, user mode
  g0[1] = lds_off;                                      // LDS byte address
  g0[2] = (unsigned)(ga & 0xffffffffu);                 // global_addr[31:0]
  g0[3] = (unsigned)((ga >> 32) & 0x01ffffffu) | (2u << 30);  // addr[56:32], type=2
  tdm_i32x8 g1;
  g1[0] = (int)(1u << 16);                              // data_size=1 (2B), mask=0
  g1[1] = (int)((nelem & 0xffffu) << 16);               // tensor_dim0[15:0] @bit48
  g1[2] = (int)(((nelem >> 16) & 0xffffu) | (1u << 16));// dim0[31:16], tensor_dim1=1
  g1[3] = (int)((nelem & 0xffffu) << 16);               // tile_dim0 @bit112
  g1[4] = 1;                                            // tile_dim1=1
  g1[5] = (int)nelem;                                   // tensor_dim0_stride[31:0]
  g1[6] = 0;
  g1[7] = 0;
  tdm_i32x4 z4 = {0, 0, 0, 0};
#if __clang_major__ >= 23
  tdm_i32x8 z8 = {0, 0, 0, 0, 0, 0, 0, 0};
  __builtin_amdgcn_tensor_load_to_lds(g0, g1, z4, z4, z8, 0);
#else
  __builtin_amdgcn_tensor_load_to_lds(g0, g1, z4, z4, 0);
#endif
}
#endif

// ---------------- weight packing ----------------
__global__ __launch_bounds__(256) void k_cvt_bf16(const float* __restrict__ src,
                                                  unsigned short* __restrict__ dst, int n) {
  int i = blockIdx.x * 256 + threadIdx.x;
  if (i < n) dst[i] = f2bf(src[i]);
}

// W[256][512][3][3] -> dst[256][2304] with k = tap*256 + c  (c<256; zero-half dropped)
__global__ __launch_bounds__(256) void k_pack_w3x3(const float* __restrict__ Wsrc,
                                                   unsigned short* __restrict__ dst) {
  int i = blockIdx.x * 256 + threadIdx.x;
  if (i >= 256 * K3_) return;
  int r = i / K3_, k = i - r * K3_;
  int tap = k >> 8, c = k & 255;
  dst[i] = f2bf(Wsrc[(size_t)(r * 512 + c) * 9 + tap]);
}

// ---------------- 1: proj_x + tanh, output NHWC bf16 ----------------
__global__ __launch_bounds__(128) void k_proj_x(const float* __restrict__ x,
                                                const unsigned short* __restrict__ Wx,
                                                const float* __restrict__ bx,
                                                unsigned short* __restrict__ xt_nhwc) {
  __shared__ __align__(16) unsigned short sA[64 * 32];
  __shared__ __align__(16) unsigned short sB[16 * 32];
  const int n = blockIdx.z, p0 = blockIdx.x * 16, m0 = blockIdx.y * 64;
  const int t = threadIdx.x, wave = t >> 5;
  const float* xb = x + (size_t)n * C_ * HW_;
  v8f acc = {};
  for (int k0 = 0; k0 < C_; k0 += 32) {
    {   // stage A: 64 rows x 32 K of W_x
      int r = t >> 1, hh = (t & 1) * 16;
      const unsigned short* src = Wx + (size_t)(m0 + r) * C_ + k0 + hh;
      *(uint4*)(sA + r * 32 + hh)     = *(const uint4*)(src);
      *(uint4*)(sA + r * 32 + hh + 8) = *(const uint4*)(src + 8);
    }
    {   // stage B: x [c][p] f32 -> LDS [tok][c] bf16, packed u32 stores
      int c0 = (t & 15) * 2, tok0 = (t >> 4) * 2;
      const float* s0 = xb + (size_t)(k0 + c0) * HW_ + p0 + tok0;
      float2 va = *(const float2*)(s0);        // channel c0, tokens tok0..tok0+1
      float2 vb2 = *(const float2*)(s0 + HW_); // channel c0+1
      *(unsigned*)(sB + (tok0 + 0) * 32 + c0) = pk2(va.x, vb2.x);
      *(unsigned*)(sB + (tok0 + 1) * 32 + c0) = pk2(va.y, vb2.y);
    }
    __syncthreads();
    v16bf fa = frag_a_lds(sA + wave * 16 * 32);
    v16bf fb = frag_b_lds(sB);
    acc = wmma_bf16(fa, fb, acc);
    __syncthreads();
  }
  int l = t & 31, col = l & 15, hf = l >> 4;
  int p = p0 + col;
  int co0 = m0 + wave * 16 + 8 * hf;          // 8 consecutive channels
  v8f tv;
#pragma unroll
  for (int j = 0; j < 8; ++j) tv[j] = tanh_fast(acc[j] + bx[co0 + j]);
  *(uint4*)(xt_nhwc + ((size_t)n * HW_ + p) * C_ + co0) = pk8(tv, 1.0f);
}

// ---------------- 2: 3x3 convs as implicit GEMM ----------------
// 2M x 2N register tile per wave: block tile 128 rows x 32 tokens, 4 waves,
// 4 WMMAs per wave per K-step against 8 b128 fragment loads.
// MODE 0: q, SAME, f32 out [n][256][1024]
// MODE 1: k, VALID, f32 out [n][256][900]
// MODE 2: v, VALID, bf16 out [n][8][32][928] (channel-major, padded)
template <int MODE>
__global__ __launch_bounds__(128) void k_conv3x3(const unsigned short* __restrict__ Wpk,
                                                 const unsigned short* __restrict__ xt,
                                                 float* __restrict__ out_f32,
                                                 unsigned short* __restrict__ out_v) {
  __shared__ __align__(16) unsigned short sA[128 * 32];
  __shared__ __align__(16) unsigned short sB[32 * 32];
  const int n = blockIdx.z, p0 = blockIdx.x * 32, m0 = blockIdx.y * 128;
  const int t = threadIdx.x, wave = t >> 5;
  const unsigned short* xb = xt + (size_t)n * HW_ * C_;
  v8f acc00 = {}, acc01 = {}, acc10 = {}, acc11 = {};   // [mi][ni]
  for (int k0 = 0; k0 < K3_; k0 += 32) {
    const int tap = k0 >> 8, cb = k0 & 255;
    const int ky = tap / 3, kx = tap - ky * 3;
    {   // stage A: 128 rows x 32 K; one 64B row per thread (+ prefetch next chunk)
      const unsigned short* src = Wpk + (size_t)(m0 + t) * K3_ + k0;
      *(uint4*)(sA + t * 32)     = *(const uint4*)(src);
      *(uint4*)(sA + t * 32 + 8) = *(const uint4*)(src + 8);
      if (k0 + 32 < K3_) __builtin_prefetch(src + 32, 0, 0);
    }
    {   // im2col B tile [32 tok][32 c]; one 16B NHWC load per thread
      int e = t * 8, c0 = e & 31, tok = e >> 5;
      int p = p0 + tok;
      uint4 val = {0, 0, 0, 0};
      if (MODE == 0) {
        int y = p >> 5, xx = p & 31;
        int iy = y + ky - 1, ix = xx + kx - 1;
        if (iy >= 0 && iy < H_ && ix >= 0 && ix < W_)
          val = *(const uint4*)(xb + ((size_t)iy * W_ + ix) * C_ + cb + c0);
      } else {
        if (p > DV_ - 1) p = DV_ - 1;       // ragged tail: duplicate, masked on store
        int y = p / 30, xx = p - y * 30;
        val = *(const uint4*)(xb + ((size_t)(y + ky) * W_ + (xx + kx)) * C_ + cb + c0);
      }
      *(uint4*)(sB + tok * 32 + c0) = val;
    }
    __syncthreads();
    v16bf fa0 = frag_a_lds(sA + (wave * 32) * 32);
    v16bf fa1 = frag_a_lds(sA + (wave * 32 + 16) * 32);
    v16bf fb0 = frag_b_lds(sB);
    v16bf fb1 = frag_b_lds(sB + 16 * 32);
    acc00 = wmma_bf16(fa0, fb0, acc00);
    acc01 = wmma_bf16(fa0, fb1, acc01);
    acc10 = wmma_bf16(fa1, fb0, acc10);
    acc11 = wmma_bf16(fa1, fb1, acc11);
    __syncthreads();
  }
  int l = t & 31, col = l & 15, hf = l >> 4;
#pragma unroll
  for (int mi = 0; mi < 2; ++mi) {
#pragma unroll
    for (int ni = 0; ni < 2; ++ni) {
      v8f acc = mi ? (ni ? acc11 : acc10) : (ni ? acc01 : acc00);
      int p = p0 + ni * 16 + col;
#pragma unroll
      for (int j = 0; j < 8; ++j) {
        int co = m0 + wave * 32 + mi * 16 + j + 8 * hf;
        if (MODE == 0) {
          out_f32[((size_t)n * C_ + co) * HW_ + p] = acc[j];
        } else if (MODE == 1) {
          if (p < DV_) out_f32[((size_t)n * C_ + co) * DV_ + p] = acc[j];
        } else {
          if (p < DV_) {
            int hd = co >> 5, cl = co & 31;
            out_v[(((size_t)n * HEADS_ + hd) * HC_ + cl) * DPAD_ + p] = f2bf(acc[j]);
          }
        }
      }
    }
  }
}

// ---------------- 3: channel-norm for q and k ----------------
__global__ __launch_bounds__(256) void k_norm_q(const float* __restrict__ qraw,
                                                unsigned short* __restrict__ qn) {
  int wid = blockIdx.x * 8 + (threadIdx.x >> 5);          // one wave per token
  int lane = threadIdx.x & 31;
  int n = wid >> 13;
  int r = wid & 8191;
  int h = r >> 10;
  int q = r & 1023;
  float v = qraw[((size_t)n * C_ + h * HC_ + lane) * HW_ + q];
  float s = v * v;
#pragma unroll
  for (int m = 1; m < 32; m <<= 1) s += __shfl_xor(s, m, 32);
  float inv = rsqrtf(fmaxf(s, 1e-24f));
  qn[(((size_t)n * HEADS_ + h) * HW_ + q) * HC_ + lane] = f2bf(v * inv);
}

__global__ __launch_bounds__(256) void k_norm_k(const float* __restrict__ kraw,
                                                const float* __restrict__ scales,
                                                unsigned short* __restrict__ kn) {
  int wid = blockIdx.x * 8 + (threadIdx.x >> 5);
  if (wid >= N_ * HEADS_ * DV_) return;
  int lane = threadIdx.x & 31;
  int n = wid / (HEADS_ * DV_);
  int r = wid - n * (HEADS_ * DV_);
  int h = r / DV_;
  int d = r - h * DV_;
  float v = kraw[((size_t)n * C_ + h * HC_ + lane) * DV_ + d];
  float s = v * v;
#pragma unroll
  for (int m = 1; m < 32; m <<= 1) s += __shfl_xor(s, m, 32);
  float inv = rsqrtf(fmaxf(s, 1e-24f));
  kn[(((size_t)n * HEADS_ + h) * DPAD_ + d) * HC_ + lane] =
      f2bf(v * inv * scales[h * HC_ + lane]);
}

// ---------------- 4: flash attention ----------------
// 4 waves / block share TDM-staged K-hat + V panels (whole head) in LDS;
// each wave owns 16 queries; inner loop is LDS + WMMA only, no barriers.
static constexpr int ATTN_PANEL = DPAD_ * HC_;                       // 29696 elems
static constexpr int ATTN_LDS_B = (2 * ATTN_PANEL + 4 * 16 * 32 * 2) * 2 + 4 * 32 * 4;

__global__ __launch_bounds__(128) void k_attn(const unsigned short* __restrict__ qn,
                                              const unsigned short* __restrict__ kn,
                                              const unsigned short* __restrict__ vb,
                                              unsigned short* __restrict__ a_buf) {
  extern __shared__ __align__(16) unsigned short smem[];
  unsigned short* sKp = smem;                        // [928][32] K-hat panel
  unsigned short* sVp = smem + ATTN_PANEL;           // [32][928] V panel
  unsigned short* sQ  = smem + 2 * ATTN_PANEL;       // 4 x [16][32]
  unsigned short* sW  = sQ + 4 * 16 * 32;            // 4 x [16][32]
  float* sS = (float*)(sW + 4 * 16 * 32);            // 4 x (16 scale + 16 L)
  const int n = blockIdx.z, hd = blockIdx.y;
  const int t = threadIdx.x, wave = t >> 5, l = t & 31;
  const int q0 = blockIdx.x * 64 + wave * 16;
  const int col = l & 15, hf = l >> 4;
  unsigned short* sQw = sQ + wave * 16 * 32;
  unsigned short* sWw = sW + wave * 16 * 32;
  float* sScale = sS + wave * 32;
  float* sL = sScale + 16;
  const unsigned short* qb = qn + (((size_t)n * HEADS_ + hd) * HW_ + q0) * HC_;
  const unsigned short* kb = kn + ((size_t)n * HEADS_ + hd) * (size_t)DPAD_ * HC_;
  const unsigned short* vB = vb + ((size_t)n * HEADS_ + hd) * (size_t)HC_ * DPAD_;

#if HAVE_TDM
  if (wave == 0) {          // one DMA per panel; EXEC is ignored by TENSOR ops
    tdm_load_1d((unsigned)(uintptr_t)sKp, kb, ATTN_PANEL);
    tdm_load_1d((unsigned)(uintptr_t)sVp, vB, ATTN_PANEL);
    __builtin_amdgcn_s_wait_tensorcnt(0);
  }
#else
  {   // fallback: cooperative vector copy
    const uint4* gK = (const uint4*)kb;
    const uint4* gV = (const uint4*)vB;
    uint4* dK = (uint4*)sKp;
    uint4* dV = (uint4*)sVp;
    const int n16 = ATTN_PANEL / 8;
    for (int i = t; i < n16; i += 128) { dK[i] = gK[i]; dV[i] = gV[i]; }
  }
#endif
  {   // per-wave q-hat tile [16 q][32 c]
    int r = l >> 1, hh = (l & 1) * 16;
    *(uint4*)(sQw + r * 32 + hh)     = *(const uint4*)(qb + r * 32 + hh);
    *(uint4*)(sQw + r * 32 + hh + 8) = *(const uint4*)(qb + r * 32 + hh + 8);
  }
  __syncthreads();          // panels + Q visible to all waves

  v16bf fq = frag_a_lds(sQw);
  v8f acc0 = {}, acc1 = {};          // a rows c=0..15 and c=16..31
  float mrun[8], lrun[8];
#pragma unroll
  for (int j = 0; j < 8; ++j) { mrun[j] = -1e30f; lrun[j] = 0.f; }

  for (int d0 = 0; d0 < DPAD_; d0 += 32) {
    v16bf fb0 = frag_b_lds(sKp + (size_t)d0 * 32);
    v16bf fb1 = frag_b_lds(sKp + (size_t)(d0 + 16) * 32);
    v8f zz = {};
    v8f lg0 = wmma_bf16(fq, fb0, zz);     // logits[q][d0..d0+15]
    v8f lg1 = wmma_bf16(fq, fb1, zz);     // logits[q][d0+16..d0+31]
    bool mA = (d0 + col)      >= DV_;
    bool mB = (d0 + 16 + col) >= DV_;
    float sc[8];
#pragma unroll
    for (int j = 0; j < 8; ++j) {         // online softmax per row q = j + 8*hf
      float a0 = mA ? -1e30f : lg0[j];
      float a1 = mB ? -1e30f : lg1[j];
      float mt = fmaxf(a0, a1);
#pragma unroll
      for (int m = 1; m < 16; m <<= 1) mt = fmaxf(mt, __shfl_xor(mt, m, 32));
      float mnew = fmaxf(mrun[j], mt);
      float w0 = __expf(a0 - mnew);
      float w1 = __expf(a1 - mnew);
      float ps = w0 + w1;
#pragma unroll
      for (int m = 1; m < 16; m <<= 1) ps += __shfl_xor(ps, m, 32);
      float f = __expf(mrun[j] - mnew);
      lrun[j] = lrun[j] * f + ps;
      mrun[j] = mnew;
      sc[j]   = f;
      sWw[(j + 8 * hf) * 32 + col]      = f2bf(w0);
      sWw[(j + 8 * hf) * 32 + 16 + col] = f2bf(w1);
    }
    if (col == 0) {
#pragma unroll
      for (int j = 0; j < 8; ++j) sScale[j + 8 * hf] = sc[j];
    }
    // wave-private LDS: same-wave DS ops are ordered, no barrier needed
    float f = sScale[col];                // accumulator cols are q = lane%16
#pragma unroll
    for (int j = 0; j < 8; ++j) { acc0[j] *= f; acc1[j] *= f; }
    v16bf fw  = frag_b_lds(sWw);                              // [K=d][N=q]
    v16bf fv0 = frag_a_lds_s(sVp + d0, DPAD_);                // v rows c=0..15
    v16bf fv1 = frag_a_lds_s(sVp + 16 * DPAD_ + d0, DPAD_);   // v rows c=16..31
    acc0 = wmma_bf16(fv0, fw, acc0);
    acc1 = wmma_bf16(fv1, fw, acc1);
  }
  if (col == 0) {
#pragma unroll
    for (int j = 0; j < 8; ++j) sL[j + 8 * hf] = lrun[j];
  }
  float invl = 1.0f / sL[col];
  // token-major a: [n][q][256]; 8 consecutive channels per lane -> b128 stores
  unsigned short* ab =
      a_buf + ((size_t)n * HW_ + q0 + col) * C_ + hd * HC_ + 8 * hf;
  *(uint4*)(ab)      = pk8(acc0, invl);
  *(uint4*)(ab + 16) = pk8(acc1, invl);
}

// ---------------- 5a: GRU gate (two GEMMs share B), h token-major bf16 ----------------
__global__ __launch_bounds__(128) void k_gate(const unsigned short* __restrict__ Wz,
                                              const unsigned short* __restrict__ Wh,
                                              const unsigned short* __restrict__ a_buf,
                                              unsigned short* __restrict__ h_tok) {
  __shared__ __align__(16) unsigned short sAz[64 * 32];
  __shared__ __align__(16) unsigned short sAh[64 * 32];
  __shared__ __align__(16) unsigned short sB[16 * 32];
  const int n = blockIdx.z, p0 = blockIdx.x * 16, m0 = blockIdx.y * 64;
  const int t = threadIdx.x, wave = t >> 5;
  const unsigned short* ab = a_buf + (size_t)n * HW_ * C_;
  v8f accz = {}, acch = {};
  for (int k0 = 0; k0 < C_; k0 += 32) {
    {
      int r = t >> 1, hh = (t & 1) * 16;
      const unsigned short* s1 = Wz + (size_t)(m0 + r) * C_ + k0 + hh;
      *(uint4*)(sAz + r * 32 + hh)     = *(const uint4*)(s1);
      *(uint4*)(sAz + r * 32 + hh + 8) = *(const uint4*)(s1 + 8);
      const unsigned short* s2 = Wh + (size_t)(m0 + r) * C_ + k0 + hh;
      *(uint4*)(sAh + r * 32 + hh)     = *(const uint4*)(s2);
      *(uint4*)(sAh + r * 32 + hh + 8) = *(const uint4*)(s2 + 8);
    }
    {   // B from token-major a: contiguous 8B loads
      int e = t * 4, c0 = e & 31, tok = e >> 5;
      unsigned long long v4 =
          *(const unsigned long long*)(ab + (size_t)(p0 + tok) * C_ + k0 + c0);
      *(unsigned long long*)(sB + tok * 32 + c0) = v4;
    }
    __syncthreads();
    v16bf faz = frag_a_lds(sAz + wave * 16 * 32);
    v16bf fah = frag_a_lds(sAh + wave * 16 * 32);
    v16bf fb  = frag_b_lds(sB);
    accz = wmma_bf16(faz, fb, accz);
    acch = wmma_bf16(fah, fb, acch);
    __syncthreads();
  }
  int l = t & 31, col = l & 15, hf = l >> 4;
  int p = p0 + col;
  int co0 = m0 + wave * 16 + 8 * hf;
  v8f hv;
#pragma unroll
  for (int j = 0; j < 8; ++j) {
    float z  = sigm_fast(accz[j] + 1.0f);     // INIT_GATE = 1.0
    hv[j] = (1.0f - z) * tanh_fast(acch[j]);  // h0 == 0
  }
  *(uint4*)(h_tok + ((size_t)n * HW_ + p) * C_ + co0) = pk8(hv, 1.0f);
}

// ---------------- 5b: out = W_out * [xt ; h] + b_out ----------------
__global__ __launch_bounds__(128) void k_out(const unsigned short* __restrict__ Wo,
                                             const unsigned short* __restrict__ xt_nhwc,
                                             const unsigned short* __restrict__ h_tok,
                                             const float* __restrict__ b_out,
                                             float* __restrict__ out) {
  __shared__ __align__(16) unsigned short sA[64 * 32];
  __shared__ __align__(16) unsigned short sB[16 * 32];
  const int n = blockIdx.z, p0 = blockIdx.x * 16, m0 = blockIdx.y * 64;
  const int t = threadIdx.x, wave = t >> 5;
  v8f acc = {};
  for (int k0 = 0; k0 < 512; k0 += 32) {
    {
      int r = t >> 1, hh = (t & 1) * 16;
      const unsigned short* src = Wo + (size_t)(m0 + r) * 512 + k0 + hh;
      *(uint4*)(sA + r * 32 + hh)     = *(const uint4*)(src);
      *(uint4*)(sA + r * 32 + hh + 8) = *(const uint4*)(src + 8);
    }
    {
      const unsigned short* base =
          (k0 < 256 ? xt_nhwc : h_tok) + (size_t)n * HW_ * C_;
      int cb = k0 & 255;
      int e = t * 4, c0 = e & 31, tok = e >> 5;
      unsigned long long v4 =
          *(const unsigned long long*)(base + (size_t)(p0 + tok) * C_ + cb + c0);
      *(unsigned long long*)(sB + tok * 32 + c0) = v4;
    }
    __syncthreads();
    v16bf fa = frag_a_lds(sA + wave * 16 * 32);
    v16bf fb = frag_b_lds(sB);
    acc = wmma_bf16(fa, fb, acc);
    __syncthreads();
  }
  int l = t & 31, col = l & 15, hf = l >> 4;
  int p = p0 + col;
#pragma unroll
  for (int j = 0; j < 8; ++j) {
    int co = m0 + wave * 16 + j + 8 * hf;
    out[((size_t)n * C_ + co) * HW_ + p] = acc[j] + b_out[co];
  }
}

// ---------------- host driver ----------------
extern "C" void kernel_launch(void* const* d_in, const int* in_sizes, int n_in,
                              void* d_out, int out_size, void* d_ws, size_t ws_size,
                              hipStream_t stream) {
  (void)in_sizes; (void)n_in; (void)out_size; (void)ws_size;
  const float* x      = (const float*)d_in[0];
  const float* W_x    = (const float*)d_in[1];
  const float* b_x    = (const float*)d_in[2];
  const float* W_q    = (const float*)d_in[3];
  const float* W_k    = (const float*)d_in[4];
  const float* W_v    = (const float*)d_in[5];
  const float* scales = (const float*)d_in[6];
  const float* W_z    = (const float*)d_in[7];
  const float* W_h    = (const float*)d_in[8];
  const float* W_out  = (const float*)d_in[9];
  const float* b_out  = (const float*)d_in[10];
  float* out = (float*)d_out;

  char* ws = (char*)d_ws;
  size_t off = 0;
  auto alloc = [&](size_t bytes) -> char* {
    char* p = ws + off;
    off += (bytes + 255) & ~(size_t)255;
    return p;
  };
  unsigned short* Wx_bf = (unsigned short*)alloc((size_t)256 * 256 * 2);
  unsigned short* Wq_bf = (unsigned short*)alloc((size_t)256 * K3_ * 2);
  unsigned short* Wk_bf = (unsigned short*)alloc((size_t)256 * K3_ * 2);
  unsigned short* Wv_bf = (unsigned short*)alloc((size_t)256 * K3_ * 2);
  unsigned short* Wz_bf = (unsigned short*)alloc((size_t)256 * 256 * 2);
  unsigned short* Wh_bf = (unsigned short*)alloc((size_t)256 * 256 * 2);
  unsigned short* Wo_bf = (unsigned short*)alloc((size_t)256 * 512 * 2);
  unsigned short* xt    = (unsigned short*)alloc((size_t)N_ * HW_ * C_ * 2);
  float*          qraw  = (float*)alloc((size_t)N_ * C_ * HW_ * 4);
  float*          kraw  = (float*)alloc((size_t)N_ * C_ * DV_ * 4);
  unsigned short* qn    = (unsigned short*)alloc((size_t)N_ * HEADS_ * HW_ * HC_ * 2);
  unsigned short* kn    = (unsigned short*)alloc((size_t)N_ * HEADS_ * DPAD_ * HC_ * 2);
  size_t vb_bytes = (size_t)N_ * HEADS_ * HC_ * DPAD_ * 2;
  unsigned short* vb    = (unsigned short*)alloc(vb_bytes);
  unsigned short* a_buf = (unsigned short*)alloc((size_t)N_ * HW_ * C_ * 2);
  unsigned short* h_tok = (unsigned short*)alloc((size_t)N_ * HW_ * C_ * 2);

  // --- weight packing ---
  k_cvt_bf16<<<256, 256, 0, stream>>>(W_x, Wx_bf, 256 * 256);
  k_cvt_bf16<<<256, 256, 0, stream>>>(W_z, Wz_bf, 256 * 256);
  k_cvt_bf16<<<256, 256, 0, stream>>>(W_h, Wh_bf, 256 * 256);
  k_cvt_bf16<<<512, 256, 0, stream>>>(W_out, Wo_bf, 256 * 512);
  k_pack_w3x3<<<2304, 256, 0, stream>>>(W_q, Wq_bf);
  k_pack_w3x3<<<2304, 256, 0, stream>>>(W_k, Wk_bf);
  k_pack_w3x3<<<2304, 256, 0, stream>>>(W_v, Wv_bf);
  hipMemsetAsync(vb, 0, vb_bytes, stream);   // zero pad region d=900..927

  // --- 1: proj_x + tanh -> NHWC bf16 ---
  k_proj_x<<<dim3(64, 4, 16), 128, 0, stream>>>(x, Wx_bf, b_x, xt);

  // --- 2: 3x3 convs (2M x 2N per wave) ---
  k_conv3x3<0><<<dim3(32, 2, 16), 128, 0, stream>>>(Wq_bf, xt, qraw, nullptr);
  k_conv3x3<1><<<dim3(29, 2, 16), 128, 0, stream>>>(Wk_bf, xt, kraw, nullptr);
  k_conv3x3<2><<<dim3(29, 2, 16), 128, 0, stream>>>(Wv_bf, xt, nullptr, vb);

  // --- 3: normalize q, k ---
  k_norm_q<<<16384, 256, 0, stream>>>(qraw, qn);                 // 16*8*1024 waves
  k_norm_k<<<14400, 256, 0, stream>>>(kraw, scales, kn);         // 16*8*900 waves

  // --- 4: flash attention (TDM-staged K/V panels, ~124 KB dynamic LDS) ---
  k_attn<<<dim3(16, HEADS_, N_), 128, ATTN_LDS_B, stream>>>(qn, kn, vb, a_buf);

  // --- 5: gate + output projection ---
  k_gate<<<dim3(64, 4, 16), 128, 0, stream>>>(Wz_bf, Wh_bf, a_buf, h_tok);
  k_out<<<dim3(64, 4, 16), 128, 0, stream>>>(Wo_bf, xt, h_tok, b_out, out);
}
